// NodeClassificationMPNTypeConstrained_59064390255319
// MI455X (gfx1250) — compile-verified, matrix operand order checked by reading, not run on hardware
//
#include <hip/hip_runtime.h>
#include <math.h>

#define DEV __device__ __forceinline__

typedef __attribute__((ext_vector_type(16))) __bf16 bv16;
typedef __attribute__((ext_vector_type(8)))  __bf16 bv8;
typedef __attribute__((ext_vector_type(8)))  float  fv8;

DEV fv8 wmma_bf16(bv16 a, bv16 b, fv8 c) {
  return __builtin_amdgcn_wmma_f32_16x16x32_bf16(false, a, false, b, (short)0, c, false, false);
}

DEV fv8 bcast8(float b) {
  fv8 v;
#pragma unroll
  for (int i = 0; i < 8; i++) v[i] = b;
  return v;
}

DEV float fadd_atomic(float* p, float v) {
  return __hip_atomic_fetch_add(p, v, __ATOMIC_RELAXED, __HIP_MEMORY_SCOPE_AGENT);
}

// Build a 16x32 bf16 A-fragment from a row-major f32 row.
// Lane layout (wave32): lanes 0-15 = rows M, hf=lane>>4 selects K sub-chunks:
//   elems [0..7]  = row[k0 + 8*hf + 0..7]
//   elems [8..15] = row[k0 + 16 + 8*hf + 0..7]
DEV bv16 a_frag_f32(const float* row, int k0, int hf) {
  const float4* p = (const float4*)(row + k0 + hf * 8);
  const float4* q = (const float4*)(row + k0 + 16 + hf * 8);
  float4 p0 = p[0], p1 = p[1], q0 = q[0], q1 = q[1];
  bv16 a;
  a[0] = (__bf16)p0.x; a[1] = (__bf16)p0.y; a[2] = (__bf16)p0.z; a[3] = (__bf16)p0.w;
  a[4] = (__bf16)p1.x; a[5] = (__bf16)p1.y; a[6] = (__bf16)p1.z; a[7] = (__bf16)p1.w;
  a[8] = (__bf16)q0.x; a[9] = (__bf16)q0.y; a[10] = (__bf16)q0.z; a[11] = (__bf16)q0.w;
  a[12] = (__bf16)q1.x; a[13] = (__bf16)q1.y; a[14] = (__bf16)q1.z; a[15] = (__bf16)q1.w;
  return a;
}

DEV bv16 a_frag_bf16(const __bf16* row, int k0, int hf) {
  bv8 x = *(const bv8*)(row + k0 + hf * 8);
  bv8 y = *(const bv8*)(row + k0 + 16 + hf * 8);
  bv16 a;
#pragma unroll
  for (int i = 0; i < 8; i++) { a[i] = x[i]; a[8 + i] = y[i]; }
  return a;
}

// Packed B fragments: frag = kb*(Np/16)+nb, each frag = 32 lanes x 16 bf16 (512 elems).
DEV bv16 b_frag(const __bf16* pk, int frag, int lane) {
  return *(const bv16*)(pk + ((size_t)frag << 9) + (lane << 4));
}

// ---- weight packer: W[Kreal,Nreal] f32 row-major -> bf16 B-fragment layout, zero padded
__global__ void pack_b_k(const float* __restrict__ W, __bf16* __restrict__ out,
                         int Kreal, int Nreal, int Kp, int Np) {
  int idx = blockIdx.x * blockDim.x + threadIdx.x;
  int total = (Kp / 32) * (Np / 16) * 512;
  if (idx >= total) return;
  int j = idx & 15;
  int lane = (idx >> 4) & 31;
  int f = idx >> 9;
  int nbc = Np / 16;
  int nb = f % nbc, kb = f / nbc;
  int n = lane & 15, hf = lane >> 4;
  int kl = (j < 8) ? (8 * hf + j) : (16 + 8 * hf + (j - 8));
  int k = kb * 32 + kl;
  int c = nb * 16 + n;
  float v = (k < Kreal && c < Nreal) ? W[(size_t)k * Nreal + c] : 0.0f;
  out[idx] = (__bf16)v;
}

__global__ void fill_u32_k(unsigned* p, unsigned v, long long n) {
  long long i = (long long)blockIdx.x * blockDim.x + threadIdx.x;
  if (i < n) p[i] = v;
}

// ---- fused 2-layer MLP over rows of X[R,64]: relu(X@W1+b1)@W2+b2
// MODE: 0 = write [R,64] f32; 1 = write col0; 2 = sigmoid(col0); 3 = 17-class + argmax
template <int N2P, int MODE>
__global__ __launch_bounds__(128)
__attribute__((amdgpu_waves_per_eu(2)))
void mlp2_wmma_k(int R, const float* X,
                 const __bf16* __restrict__ w1p, const float* __restrict__ b1,
                 const __bf16* __restrict__ w2p, const float* __restrict__ b2,
                 int n2real, float* out0, int* out1) {
  constexpr int NBC = N2P / 16;
  __shared__ __align__(32) char sm[4][4096];
  int wv = threadIdx.x >> 5, lane = threadIdx.x & 31;
  int m = lane & 15, hf = lane >> 4, n = m;
  int gw = blockIdx.x * (blockDim.x >> 5) + wv;
  int nw = gridDim.x * (blockDim.x >> 5);
  int tiles = (R + 15) >> 4;
  __bf16* hb = (__bf16*)sm[wv];

  // deliberate register-resident B fragments (loop-invariant)
  bv16 B1[8];
#pragma unroll
  for (int f = 0; f < 8; f++) B1[f] = b_frag(w1p, f, lane);
  bv16 B2[2 * NBC];
#pragma unroll
  for (int f = 0; f < 2 * NBC; f++) B2[f] = b_frag(w2p, f, lane);

  float bias1[4];
#pragma unroll
  for (int nb = 0; nb < 4; nb++) bias1[nb] = b1[nb * 16 + n];
  float bias2[NBC];
#pragma unroll
  for (int nb = 0; nb < NBC; nb++) {
    int col = nb * 16 + n;
    bias2[nb] = (col < n2real) ? b2[col] : 0.0f;
  }

  for (int t = gw; t < tiles; t += nw) {
    int row = t * 16 + m;
    int rc = row < R ? row : R - 1;
    const float* xr = X + (size_t)rc * 64;
    bv16 a0 = a_frag_f32(xr, 0, hf);
    bv16 a1 = a_frag_f32(xr, 32, hf);

    // layer 1: 64 -> 64, relu, stage as bf16 rows in LDS
#pragma unroll
    for (int nb = 0; nb < 4; nb++) {
      fv8 acc = bcast8(bias1[nb]);
      acc = wmma_bf16(a0, B1[0 * 4 + nb], acc);
      acc = wmma_bf16(a1, B1[1 * 4 + nb], acc);
#pragma unroll
      for (int v = 0; v < 8; v++) {
        float f = fmaxf(acc[v], 0.0f);
        hb[(v + hf * 8) * 64 + nb * 16 + n] = (__bf16)f;
      }
    }
    __builtin_amdgcn_wave_barrier();

    const __bf16* hr = hb + m * 64;
    bv16 c0 = a_frag_bf16(hr, 0, hf);
    bv16 c1 = a_frag_bf16(hr, 32, hf);

    float* cs = (float*)sm[wv];
#pragma unroll
    for (int nb = 0; nb < NBC; nb++) {
      int col = nb * 16 + n;
      fv8 acc = bcast8(bias2[nb]);
      acc = wmma_bf16(c0, B2[0 * NBC + nb], acc);
      acc = wmma_bf16(c1, B2[1 * NBC + nb], acc);
      if (MODE == 0) {
#pragma unroll
        for (int v = 0; v < 8; v++) {
          int r = t * 16 + v + hf * 8;
          if (r < R) out0[(size_t)r * 64 + col] = acc[v];
        }
      } else if (MODE == 1 || MODE == 2) {
        if (col == 0) {
#pragma unroll
          for (int v = 0; v < 8; v++) {
            int r = t * 16 + v + hf * 8;
            float val = acc[v];
            if (MODE == 2) val = 1.0f / (1.0f + __expf(-val));
            if (r < R) out0[r] = val;
          }
        }
      } else { // MODE 3: stage 16x32 f32 tile for class head
#pragma unroll
        for (int v = 0; v < 8; v++) cs[(v + hf * 8) * 32 + col] = acc[v];
      }
    }
    if (MODE == 3) {
      __builtin_amdgcn_wave_barrier();
      if (lane < 16) {
        int r = t * 16 + lane;
        if (r < R) {
          float best = -1e30f;
          int bi = 0;
#pragma unroll
          for (int c = 0; c < 17; c++) {
            float v = cs[lane * 32 + c];
            out0[(size_t)r * 17 + c] = v;
            if (v > best) { best = v; bi = c; }
          }
          out1[r] = bi;
        }
      }
      __builtin_amdgcn_wave_barrier();
    }
  }
}

// ---- single linear layer: out[R,64] = [s0 | s1] @ Wp + bias  (K = NKB*32)
template <int NKB>
__global__ __launch_bounds__(128)
__attribute__((amdgpu_waves_per_eu(2)))
void gemm_wmma_k(int R, const float* s0, const float* s1,
                 const __bf16* __restrict__ wp, const float* __restrict__ bias,
                 float* out) {
  int wv = threadIdx.x >> 5, lane = threadIdx.x & 31;
  int m = lane & 15, hf = lane >> 4, n = m;
  int gw = blockIdx.x * (blockDim.x >> 5) + wv;
  int nw = gridDim.x * (blockDim.x >> 5);
  int tiles = (R + 15) >> 4;

  bv16 B[NKB * 4];
#pragma unroll
  for (int f = 0; f < NKB * 4; f++) B[f] = b_frag(wp, f, lane);
  float bb[4];
#pragma unroll
  for (int nb = 0; nb < 4; nb++) bb[nb] = bias[nb * 16 + n];

  for (int t = gw; t < tiles; t += nw) {
    int row = t * 16 + m;
    int rc = row < R ? row : R - 1;
    bv16 a[NKB];
    const float* r0 = s0 + (size_t)rc * 64;
    a[0] = a_frag_f32(r0, 0, hf);
    a[1] = a_frag_f32(r0, 32, hf);
    if (NKB == 4) {
      const float* r1 = s1 + (size_t)rc * 64;
      a[2] = a_frag_f32(r1, 0, hf);
      a[3] = a_frag_f32(r1, 32, hf);
    }
#pragma unroll
    for (int nb = 0; nb < 4; nb++) {
      fv8 acc = bcast8(bb[nb]);
#pragma unroll
      for (int kb = 0; kb < NKB; kb++)
        acc = wmma_bf16(a[kb], B[kb * 4 + nb], acc);
#pragma unroll
      for (int v = 0; v < 8; v++) {
        int r = t * 16 + v + hf * 8;
        if (r < R) out[(size_t)r * 64 + nb * 16 + n] = acc[v];
      }
    }
  }
}

// ---- fused message MLP + segment-sum: per 16-edge tile
//   m = relu([nf[src]|nf[tgt]|ef] @ W1 + b1) @ W2 + b2 ; ef = m (in place) ; aggr[tgt] += m
__global__ __launch_bounds__(128)
__attribute__((amdgpu_waves_per_eu(4)))
void msg_wmma_k(int E, int Etiles, const int* __restrict__ ei,
                const float* __restrict__ nf, float* ef, float* aggr,
                const __bf16* __restrict__ wsrc,  // me_w1p (48KB) + me_w2p (16KB) contiguous
                const float* __restrict__ b1, const float* __restrict__ b2) {
  extern __shared__ __align__(32) char smraw[];
  __bf16* lw1 = (__bf16*)smraw;          // 24576 bf16
  __bf16* lw2 = lw1 + 24576;             // 8192 bf16
  char* stage = smraw + 65536;           // 4KB per wave (h bf16 16x128 / m f32 16x64)

  for (int i = threadIdx.x; i < 4096; i += blockDim.x)
    ((int4*)smraw)[i] = ((const int4*)wsrc)[i];
  __syncthreads();

  int wv = threadIdx.x >> 5, lane = threadIdx.x & 31;
  int m = lane & 15, hf = lane >> 4, n = m;
  int gw = blockIdx.x * (blockDim.x >> 5) + wv;
  int nw = gridDim.x * (blockDim.x >> 5);
  char* myst = stage + wv * 4096;
  __bf16* hb = (__bf16*)myst;
  float* mb = (float*)myst;

  float bb1[8], bb2[4];
#pragma unroll
  for (int nb = 0; nb < 8; nb++) bb1[nb] = b1[nb * 16 + n];
#pragma unroll
  for (int nb = 0; nb < 4; nb++) bb2[nb] = b2[nb * 16 + n];

  for (int t = gw; t < Etiles; t += nw) {
    int e0 = t * 16;
    int em = e0 + m; if (em >= E) em = E - 1;
    int es = ei[em], et = ei[E + em];
    const float* rs = nf + (size_t)es * 64;
    const float* rt = nf + (size_t)et * 64;
    const float* re = ef + (size_t)em * 64;
    bv16 a[6];
    a[0] = a_frag_f32(rs, 0, hf);  a[1] = a_frag_f32(rs, 32, hf);
    a[2] = a_frag_f32(rt, 0, hf);  a[3] = a_frag_f32(rt, 32, hf);
    a[4] = a_frag_f32(re, 0, hf);  a[5] = a_frag_f32(re, 32, hf);

    // layer 1: 192 -> 128, relu, stage bf16
#pragma unroll
    for (int nb = 0; nb < 8; nb++) {
      fv8 acc = bcast8(bb1[nb]);
#pragma unroll
      for (int kb = 0; kb < 6; kb++)
        acc = wmma_bf16(a[kb], b_frag(lw1, kb * 8 + nb, lane), acc);
#pragma unroll
      for (int v = 0; v < 8; v++) {
        float f = fmaxf(acc[v], 0.0f);
        hb[(v + hf * 8) * 128 + nb * 16 + n] = (__bf16)f;
      }
    }
    __builtin_amdgcn_wave_barrier();

    const __bf16* hr = hb + m * 128;
    bv16 c[4];
#pragma unroll
    for (int kb = 0; kb < 4; kb++) c[kb] = a_frag_bf16(hr, kb * 32, hf);

    // layer 2: 128 -> 64, stage f32 (reuses hb region; c[] already in regs)
#pragma unroll
    for (int nb = 0; nb < 4; nb++) {
      fv8 acc = bcast8(bb2[nb]);
#pragma unroll
      for (int kb = 0; kb < 4; kb++)
        acc = wmma_bf16(c[kb], b_frag(lw2, kb * 4 + nb, lane), acc);
#pragma unroll
      for (int v = 0; v < 8; v++) mb[(v + hf * 8) * 64 + nb * 16 + n] = acc[v];
    }
    __builtin_amdgcn_wave_barrier();

    // coalesced in-place ef write
    const float4* mb4 = (const float4*)mb;
    for (int i = lane; i < 256; i += 32) {
      int r = i >> 4, c4 = i & 15;
      int e = e0 + r;
      if (e < E) *(float4*)(ef + (size_t)e * 64 + c4 * 4) = mb4[i];
    }
    // segment-sum: aggr[tgt] += m
    for (int i = lane; i < 1024; i += 32) {
      int r = i >> 6, cc = i & 63;
      int e = e0 + r;
      if (e < E) {
        int tg = ei[E + e];
        fadd_atomic(aggr + (size_t)tg * 64 + cc, mb[i]);
      }
    }
    __builtin_amdgcn_wave_barrier();
  }
}

// ---- type-constrained scatter-softmax pieces
DEV unsigned f2key(float f) {
  unsigned b = __float_as_uint(f);
  return (b & 0x80000000u) ? ~b : (b | 0x80000000u);
}
DEV float key2f(unsigned k) {
  unsigned b = (k & 0x80000000u) ? (k & 0x7FFFFFFFu) : ~k;
  return __uint_as_float(b);
}

__global__ void score_k(int E, const int* __restrict__ ei, const float* __restrict__ emb,
                        const int* __restrict__ types, float* scores, int* seg, unsigned* mxU) {
  int e = blockIdx.x * blockDim.x + threadIdx.x;
  if (e >= E) return;
  int s = ei[e], t = ei[E + e];
  const float4* a = (const float4*)(emb + (size_t)s * 64);
  const float4* b = (const float4*)(emb + (size_t)t * 64);
  float acc = 0.0f;
#pragma unroll
  for (int i = 0; i < 16; i++) {
    float4 x = a[i], y = b[i];
    acc += x.x * y.x + x.y * y.y + x.z * y.z + x.w * y.w;
  }
  int sg = t * 17 + types[s];
  scores[e] = acc;
  seg[e] = sg;
  atomicMax(mxU + sg, f2key(acc));
}

__global__ void exp_k(int E, float* scores, const int* __restrict__ seg,
                      const unsigned* __restrict__ mxU, float* denom) {
  int e = blockIdx.x * blockDim.x + threadIdx.x;
  if (e >= E) return;
  int sg = seg[e];
  float ex = __expf(scores[e] - key2f(mxU[sg]));
  scores[e] = ex;
  fadd_atomic(denom + sg, ex);
}

__global__ void outmul_k(int E, const float* __restrict__ ex, const int* __restrict__ seg,
                         const float* __restrict__ denom, const float* __restrict__ sig,
                         float* eout) {
  int e = blockIdx.x * blockDim.x + threadIdx.x;
  if (e >= E) return;
  eout[e] = ex[e] / denom[seg[e]] * sig[e];
}

extern "C" void kernel_launch(void* const* d_in, const int* in_sizes, int n_in,
                              void* d_out, int out_size, void* d_ws, size_t ws_size,
                              hipStream_t stream) {
  (void)n_in; (void)out_size; (void)ws_size;
  const float* x      = (const float*)d_in[0];
  const float* eattr  = (const float*)d_in[1];
  const int*   ei     = (const int*)d_in[2];
  const float *ne_w1 = (const float*)d_in[4],  *ne_b1 = (const float*)d_in[5];
  const float *ne_w2 = (const float*)d_in[6],  *ne_b2 = (const float*)d_in[7];
  const float *ee_w1 = (const float*)d_in[8],  *ee_b1 = (const float*)d_in[9];
  const float *ee_w2 = (const float*)d_in[10], *ee_b2 = (const float*)d_in[11];
  const float *me_w1 = (const float*)d_in[12], *me_b1 = (const float*)d_in[13];
  const float *me_w2 = (const float*)d_in[14], *me_b2 = (const float*)d_in[15];
  const float *nu_w1 = (const float*)d_in[16], *nu_b1 = (const float*)d_in[17];
  const float *ec_w1 = (const float*)d_in[18], *ec_b1 = (const float*)d_in[19];
  const float *ec_w2 = (const float*)d_in[20], *ec_b2 = (const float*)d_in[21];
  const float *nc_w1 = (const float*)d_in[22], *nc_b1 = (const float*)d_in[23];
  const float *nc_w2 = (const float*)d_in[24], *nc_b2 = (const float*)d_in[25];
  const float *cl_w1 = (const float*)d_in[26], *cl_b1 = (const float*)d_in[27];
  const float *cl_w2 = (const float*)d_in[28], *cl_b2 = (const float*)d_in[29];
  const float *ce_w  = (const float*)d_in[30], *ce_b  = (const float*)d_in[31];

  const int N = in_sizes[0] / 64;
  const int E = in_sizes[1] / 64;

  // d_out layout: edge_out[E], node_pred[N], class_pred[17N], nf[64N], ef[64E]
  float* out       = (float*)d_out;
  float* edge_out  = out;
  float* node_pred = out + E;
  float* class_out = out + E + (size_t)N;
  float* nf        = out + E + (size_t)18 * N;
  float* ef        = nf + (size_t)64 * N;

  // workspace carving (256B aligned)
  char* w = (char*)d_ws;
  auto carve = [&](size_t bytes) -> char* {
    char* p = w; w += (bytes + 255) & ~(size_t)255; return p;
  };
  __bf16* me_pack = (__bf16*)carve(65536);            // me_w1p (48KB) + me_w2p (16KB)
  __bf16* me_w2p  = me_pack + 24576;
  __bf16* ne_w1p = (__bf16*)carve(8192);
  __bf16* ne_w2p = (__bf16*)carve(8192);
  __bf16* ee_w1p = (__bf16*)carve(8192);
  __bf16* ee_w2p = (__bf16*)carve(8192);
  __bf16* nu_w1p = (__bf16*)carve(16384);
  __bf16* ec_w1p = (__bf16*)carve(8192);
  __bf16* ec_w2p = (__bf16*)carve(2048);
  __bf16* nc_w1p = (__bf16*)carve(8192);
  __bf16* nc_w2p = (__bf16*)carve(2048);
  __bf16* cl_w1p = (__bf16*)carve(8192);
  __bf16* cl_w2p = (__bf16*)carve(4096);
  __bf16* ce_wp  = (__bf16*)carve(8192);
  float*    aggr   = (float*)carve((size_t)N * 64 * 4);
  float*    nemb   = (float*)carve((size_t)N * 64 * 4);
  float*    scores = (float*)carve((size_t)E * 4);
  int*      seg    = (int*)carve((size_t)E * 4);
  float*    sigE   = (float*)carve((size_t)E * 4);
  int*      types  = (int*)carve((size_t)N * 4);
  unsigned* mxU    = (unsigned*)carve((size_t)N * 17 * 4);
  float*    denom  = (float*)carve((size_t)N * 17 * 4);

  auto pack = [&](const float* W, __bf16* dst, int Kr, int Nr, int Kp, int Np) {
    int total = (Kp / 32) * (Np / 16) * 512;
    pack_b_k<<<(total + 255) / 256, 256, 0, stream>>>(W, dst, Kr, Nr, Kp, Np);
  };
  pack(ne_w1, ne_w1p, 64, 64, 64, 64);    pack(ne_w2, ne_w2p, 64, 64, 64, 64);
  pack(ee_w1, ee_w1p, 64, 64, 64, 64);    pack(ee_w2, ee_w2p, 64, 64, 64, 64);
  pack(me_w1, me_pack, 192, 128, 192, 128);
  pack(me_w2, me_w2p, 128, 64, 128, 64);
  pack(nu_w1, nu_w1p, 128, 64, 128, 64);
  pack(ec_w1, ec_w1p, 64, 64, 64, 64);    pack(ec_w2, ec_w2p, 64, 1, 64, 16);
  pack(nc_w1, nc_w1p, 64, 64, 64, 64);    pack(nc_w2, nc_w2p, 64, 1, 64, 16);
  pack(cl_w1, cl_w1p, 64, 64, 64, 64);    pack(cl_w2, cl_w2p, 64, 17, 64, 32);
  pack(ce_w,  ce_wp, 64, 64, 64, 64);

  const int Ntiles = (N + 15) / 16, Etiles = (E + 15) / 16;
  auto wb = [](int tiles) { int b = (tiles + 3) / 4; return b > 0 ? b : 1; };

  // encoders (write nf/ef in their d_out homes)
  mlp2_wmma_k<64, 0><<<wb(Ntiles), 128, 0, stream>>>(N, x, ne_w1p, ne_b1, ne_w2p, ne_b2,
                                                     64, nf, nullptr);
  mlp2_wmma_k<64, 0><<<wb(Etiles), 128, 0, stream>>>(E, eattr, ee_w1p, ee_b1, ee_w2p, ee_b2,
                                                     64, ef, nullptr);

  // 3 message-passing steps
  for (int step = 0; step < 3; step++) {
    long long na = (long long)N * 64;
    fill_u32_k<<<(unsigned)((na + 255) / 256), 256, 0, stream>>>((unsigned*)aggr, 0u, na);
    msg_wmma_k<<<1024, 128, 81920, stream>>>(E, Etiles, ei, nf, ef, aggr,
                                             me_pack, me_b1, me_b2);
    gemm_wmma_k<4><<<wb(Ntiles), 128, 0, stream>>>(N, nf, aggr, nu_w1p, nu_b1, nf);
  }

  // heads
  mlp2_wmma_k<16, 1><<<wb(Ntiles), 128, 0, stream>>>(N, nf, nc_w1p, nc_b1, nc_w2p, nc_b2,
                                                     1, node_pred, nullptr);
  mlp2_wmma_k<32, 3><<<wb(Ntiles), 128, 0, stream>>>(N, nf, cl_w1p, cl_b1, cl_w2p, cl_b2,
                                                     17, class_out, types);
  mlp2_wmma_k<16, 2><<<wb(Etiles), 128, 0, stream>>>(E, ef, ec_w1p, ec_b1, ec_w2p, ec_b2,
                                                     1, sigE, nullptr);
  gemm_wmma_k<2><<<wb(Ntiles), 128, 0, stream>>>(N, nf, nullptr, ce_wp, ce_b, nemb);

  // type-constrained scatter-softmax
  long long ns = (long long)N * 17;
  fill_u32_k<<<(unsigned)((ns + 255) / 256), 256, 0, stream>>>(mxU, 0u, ns);
  fill_u32_k<<<(unsigned)((ns + 255) / 256), 256, 0, stream>>>((unsigned*)denom, 0u, ns);
  score_k<<<(E + 255) / 256, 256, 0, stream>>>(E, ei, nemb, types, scores, seg, mxU);
  exp_k<<<(E + 255) / 256, 256, 0, stream>>>(E, scores, seg, mxU, denom);
  outmul_k<<<(E + 255) / 256, 256, 0, stream>>>(E, scores, seg, denom, sigE, edge_out);
}
